// LSTM_scratch_72353019069113
// MI455X (gfx1250) — matrix operand order repaired and checked by reading, not compiled, not run on hardware
//
#include <hip/hip_runtime.h>
#include <hip/hip_bf16.h>
#include <stdint.h>

#define HID    256
#define BATCH  256
#define DIN    64
#define SEQ    1024
#define NWG    16     // one workgroup per 16-wide hidden slice
#define TPB    256    // 8 wave32 per WG
#define NCHUNK 10     // K=32 chunks: 2 from x (D=64) + 8 from h (HID=256)

typedef __attribute__((ext_vector_type(16))) __bf16   v16bf;
typedef __attribute__((ext_vector_type(8)))  __bf16   v8bf;
typedef __attribute__((ext_vector_type(2)))  __bf16   v2bf;
typedef __attribute__((ext_vector_type(8)))  float    v8f;
typedef __attribute__((ext_vector_type(4)))  uint32_t u32x4;
typedef __attribute__((ext_vector_type(8)))  uint32_t u32x8;

static __device__ __forceinline__ uint32_t pack2_bf16(float a, float b) {
  union { v2bf v; uint32_t u; } c;
  c.v = (v2bf){(__bf16)a, (__bf16)b};      // hardware v_cvt_pk_bf16_f32
  return c.u;
}
static __device__ __forceinline__ uint16_t f_to_bf16(float a) {
  union { __bf16 b; uint16_t u; } c;
  c.b = (__bf16)a;
  return c.u;
}
static __device__ __forceinline__ float sigm(float x) { return 1.0f / (1.0f + __expf(-x)); }
static __device__ __forceinline__ float tanh_fast(float x) {
  x = fminf(fmaxf(x, -15.0f), 15.0f);
  float e = __expf(2.0f * x);
  return (e - 1.0f) / (e + 1.0f);
}

// Device-wide sense-reversing barrier (16 co-resident WGs).
static __device__ __forceinline__ void global_barrier(unsigned* bar, volatile unsigned* gen) {
  __syncthreads();
  if (threadIdx.x == 0) {
    __threadfence();                       // make h-slice stores visible device-wide
    unsigned g = *gen;
    if (atomicAdd(bar, 1u) == NWG - 1) {
      *(volatile unsigned*)bar = 0u;       // reset before releasing
      __threadfence();
      *gen = g + 1u;
    } else {
      while (*gen == g) { __builtin_amdgcn_s_sleep(2); }
    }
    __threadfence();
  }
  __syncthreads();
}

__global__ void lstm_init(const float* __restrict__ h0, uint16_t* __restrict__ hb0,
                          unsigned* bar, unsigned* gen) {
  int i = blockIdx.x * blockDim.x + threadIdx.x;
  if (i < BATCH * HID) hb0[i] = f_to_bf16(h0[i]);
  if (i == 0) { bar[0] = 0u; gen[0] = 0u; }
}

__global__ void __launch_bounds__(TPB)
lstm_persistent(const float* __restrict__ x,
                const float* __restrict__ c0,
                const float* __restrict__ Wi, const float* __restrict__ bi,
                const float* __restrict__ Wh, const float* __restrict__ bh,
                uint16_t* __restrict__ hb0, uint16_t* __restrict__ hb1,
                float* __restrict__ hF,
                unsigned* bar, unsigned* gen) {
  // LDS: Wh B-frags 32KB + Wi B-frags 8KB + double-buffered A chunk 32KB + biases = 72.3KB
  __shared__ __attribute__((aligned(32))) uint32_t sWhB[32 * 256];
  __shared__ __attribute__((aligned(32))) uint32_t sWiB[8 * 256];
  __shared__ __attribute__((aligned(32))) uint32_t sBuf[2 * 16 * 256]; // 2 x [256 rows][32 bf16]
  __shared__ float sGB[64];

  const int tid  = threadIdx.x;
  const int wv   = tid >> 5;          // wave 0..7
  const int lane = tid & 31;
  const int n0   = blockIdx.x * 16;   // this WG's hidden-column slice

  // ---- one-time: stage Wh/Wi column slices into LDS in WMMA B-layout (bf16) ----
  // B-layout per 32x16 tile: lane = n + 16*(k>=16); dword dw packs k = 16*(lane>=16) + {2dw, 2dw+1}
  {
    const int lane_s = tid >> 3;       // 0..31
    const int dw     = tid & 7;        // 0..7
    const int bn     = lane_s & 15;    // column within tile
    const int k0B    = ((lane_s >> 4) << 4) + 2 * dw;
    for (int tile = 0; tile < 32; ++tile) {          // Wh: tile = q*8 + ks
      int q = tile >> 3, ks = tile & 7;
      const float* pw = Wh + (size_t)(q * HID + n0 + bn) * HID + ks * 32 + k0B;
      sWhB[(tile * 32 + lane_s) * 8 + dw] = pack2_bf16(pw[0], pw[1]);
    }
    for (int tile = 0; tile < 8; ++tile) {           // Wi: tile = q*2 + ks
      int q = tile >> 1, ks = tile & 1;
      const float* pw = Wi + (size_t)(q * HID + n0 + bn) * DIN + ks * 32 + k0B;
      sWiB[(tile * 32 + lane_s) * 8 + dw] = pack2_bf16(pw[0], pw[1]);
    }
  }
  if (tid < 64) {
    int q = tid >> 4, n = tid & 15;
    sGB[tid] = bi[q * HID + n0 + n] + bh[q * HID + n0 + n];
  }
  __syncthreads();

  // ---- c-state resident in accumulator-layout VGPRs ----
  const int ln  = lane & 15;
  const int lhi = lane >> 4;
  v8f creg[2];
  #pragma unroll
  for (int mt2 = 0; mt2 < 2; ++mt2)
    #pragma unroll
    for (int j = 0; j < 8; ++j) {
      int m = (wv * 2 + mt2) * 16 + j + 8 * lhi;
      creg[mt2][j] = c0[(size_t)m * HID + n0 + ln];
    }

  const uint32_t ldsBufAddr = (uint32_t)(uintptr_t)(&sBuf[0]);  // LDS byte offset (addr[31:0])

  for (int t = 0; t < SEQ; ++t) {
    const uint16_t* cur = (t & 1) ? hb1 : hb0;   // h_t   (read)
    uint16_t*       nxt = (t & 1) ? hb0 : hb1;   // h_t+1 (write)

    // prefetch next timestep's x rows while we grind this step
    if (t + 1 < SEQ) {
      const float* pxn = x + (size_t)tid * (SEQ * DIN) + (size_t)(t + 1) * DIN;
      __builtin_prefetch(pxn, 0, 1);
      __builtin_prefetch(pxn + 32, 0, 1);
    }

    // ---- prologue: stage both x chunks (f32 -> bf16, row-major) into buf0 / buf1 ----
    #pragma unroll
    for (int cx = 0; cx < 2; ++cx) {
      uint32_t* bs = sBuf + cx * 4096;
      #pragma unroll
      for (int i = 0; i < 16; ++i) {
        int idx = i * 256 + tid;   // 4096 dwords
        int row = idx >> 4;
        int p   = idx & 15;
        const float* px = x + (size_t)row * (SEQ * DIN) + (size_t)t * DIN + cx * 32 + 2 * p;
        bs[idx] = pack2_bf16(px[0], px[1]);
      }
    }
    __syncthreads();

    v8f acc[2][4];
    #pragma unroll
    for (int a = 0; a < 2; ++a)
      #pragma unroll
      for (int b = 0; b < 4; ++b)
        acc[a][b] = (v8f){0.f, 0.f, 0.f, 0.f, 0.f, 0.f, 0.f, 0.f};

    // ---- software-pipelined chunk loop: TDM for chunk ch+2 overlaps WMMA on chunk ch+1 ----
    #pragma unroll
    for (int ch = 0; ch < NCHUNK; ++ch) {
      const uint32_t* bufc = sBuf + (ch & 1) * 4096;

      // WMMA: 2 M-tiles x 4 gate tiles per wave
      const __bf16* bufh = (const __bf16*)bufc;
      #pragma unroll
      for (int mt2 = 0; mt2 < 2; ++mt2) {
        int row = (wv * 2 + mt2) * 16 + ln;
        const __bf16* pa = bufh + row * 32 + lhi * 16;    // two contiguous 16B chunks
        v8bf alo = *(const v8bf*)(pa);
        v8bf ahi = *(const v8bf*)(pa + 8);
        v16bf a = __builtin_shufflevector(alo, ahi, 0, 1, 2, 3, 4, 5, 6, 7,
                                          8, 9, 10, 11, 12, 13, 14, 15);
        #pragma unroll
        for (int q = 0; q < 4; ++q) {
          const uint32_t* bp = (ch < 2)
              ? (sWiB + ((q * 2 + ch)       * 32 + lane) * 8)
              : (sWhB + ((q * 8 + (ch - 2)) * 32 + lane) * 8);
          const v16bf b = *(const v16bf*)bp;
          acc[mt2][q] = __builtin_amdgcn_wmma_f32_16x16x32_bf16(
              false, a, false, b, (short)0, acc[mt2][q], false, false);
        }
      }
      __syncthreads();            // everyone done reading buf[ch&1]

      if (wv == 0) {
        if (ch + 2 < NCHUNK) {
          // issue TDM for chunk ch+2 (h K-slice ks = ch) into the freed buffer:
          // 2D tile 32x256 bf16 out of the 256x256 h matrix, row stride HID.
          int ks = ch;
          uint64_t ga = (uint64_t)(uintptr_t)cur + (uint64_t)ks * 64;  // tile start (bytes)
          u32x4 g0 = { 1u,                                 // count=1, user descriptor
                       ldsBufAddr + (uint32_t)((ch & 1) * 16384),      // lds_addr
                       (uint32_t)ga,                       // global_addr[31:0]
                       ((uint32_t)(ga >> 32) & 0x01FFFFFFu) | 0x80000000u }; // addr[56:32]|type=2
          u32x8 g1 = { 0x00010000u,                        // data_size=2B, no flags
                       (uint32_t)HID << 16,                // tensor_dim0 lo16 (=256)
                       (uint32_t)BATCH << 16,              // tensor_dim0 hi | tensor_dim1 lo (=256)
                       32u << 16,                          // tensor_dim1 hi | tile_dim0=32
                       (uint32_t)BATCH,                    // tile_dim1=256, tile_dim2=0
                       (uint32_t)HID,                      // tensor_dim0_stride lo32 (=256)
                       0u, 0u };
          asm volatile("tensor_load_to_lds %0, %1" :: "s"(g0), "s"(g1) : "memory");
          // in-order TDM completion: <=1 outstanding means chunk ch+1 has landed
          __builtin_amdgcn_s_wait_tensorcnt(1);
        } else {
          __builtin_amdgcn_s_wait_tensorcnt(0);
        }
      }
      __syncthreads();            // publish chunk ch+1 to all waves
    }

    // ---- gates (f,i,g,o) + cell/state update; write h slice for next step ----
    #pragma unroll
    for (int mt2 = 0; mt2 < 2; ++mt2) {
      #pragma unroll
      for (int j = 0; j < 8; ++j) {
        float pf = acc[mt2][0][j] + sGB[ 0 + ln];
        float pi = acc[mt2][1][j] + sGB[16 + ln];
        float pg = acc[mt2][2][j] + sGB[32 + ln];
        float po = acc[mt2][3][j] + sGB[48 + ln];
        float f  = sigm(pf);
        float ii = sigm(pi);
        float g  = tanh_fast(pg);
        float o  = sigm(po);
        float c  = creg[mt2][j] * f + ii * g;
        creg[mt2][j] = c;
        float h  = tanh_fast(c) * o;
        int m = (wv * 2 + mt2) * 16 + j + 8 * lhi;
        nxt[(size_t)m * HID + n0 + ln] = f_to_bf16(h);
        if (t == SEQ - 1) hF[(size_t)m * HID + n0 + ln] = h;
      }
    }
    global_barrier(bar, gen);
  }
}

__global__ void lstm_cls(const float* __restrict__ hF,
                         const float* __restrict__ Wcls, const float* __restrict__ bcls,
                         float* __restrict__ out) {
  int i = blockIdx.x * blockDim.x + threadIdx.x;
  if (i >= BATCH * 10) return;
  int b = i / 10, j = i % 10;
  const float* hr = hF   + (size_t)b * HID;
  const float* wr = Wcls + (size_t)j * HID;
  float s = bcls[j];
  for (int k = 0; k < HID; ++k) s += hr[k] * wr[k];
  out[i] = s;
}

extern "C" void kernel_launch(void* const* d_in, const int* in_sizes, int n_in,
                              void* d_out, int out_size, void* d_ws, size_t ws_size,
                              hipStream_t stream) {
  const float* x    = (const float*)d_in[0];
  const float* h0   = (const float*)d_in[1];
  const float* c0   = (const float*)d_in[2];
  const float* Wi   = (const float*)d_in[3];
  const float* bi   = (const float*)d_in[4];
  const float* Wh   = (const float*)d_in[5];
  const float* bh   = (const float*)d_in[6];
  const float* Wcls = (const float*)d_in[7];
  const float* bcls = (const float*)d_in[8];
  (void)in_sizes; (void)n_in; (void)out_size; (void)ws_size;

  uint8_t* ws = (uint8_t*)d_ws;
  unsigned* bar = (unsigned*)(ws + 0);
  unsigned* gen = (unsigned*)(ws + 128);
  uint16_t* hb0 = (uint16_t*)(ws + 256);
  uint16_t* hb1 = (uint16_t*)(ws + 256 + (size_t)BATCH * HID * 2);
  float*    hF  = (float*)   (ws + 256 + (size_t)BATCH * HID * 4);

  lstm_init<<<(BATCH * HID + TPB - 1) / TPB, TPB, 0, stream>>>(h0, hb0, bar, gen);
  lstm_persistent<<<NWG, TPB, 0, stream>>>(x, c0, Wi, bi, Wh, bh, hb0, hb1, hF, bar, gen);
  lstm_cls<<<(BATCH * 10 + TPB - 1) / TPB, TPB, 0, stream>>>(hF, Wcls, bcls, (float*)d_out);
}
